// BatchedGraphTemporalFourierLayer_56427280335234
// MI455X (gfx1250) — compile-verified, matrix operand order checked by reading, not run on hardware
//
#include <hip/hip_runtime.h>
#include <hip/hip_bf16.h>
#include <math.h>

// ---- problem constants (match reference) ----
#define BB 8
#define CC 32
#define NN 1024
#define TT 64
#define KNE 8
#define MS 16
#define MT 4

typedef __attribute__((ext_vector_type(16))) _Float16 v16h;
typedef __attribute__((ext_vector_type(8)))  float    v8f;
typedef __attribute__((ext_vector_type(4)))  unsigned int u32x4;
typedef __attribute__((ext_vector_type(4)))  int      i32x4;
typedef __attribute__((ext_vector_type(8)))  int      i32x8;

__device__ __forceinline__ v8f wmma16(v16h a, v16h b, v8f c) {
    return __builtin_amdgcn_wmma_f32_16x16x32_f16(
        /*neg_a=*/false, a, /*neg_b=*/false, b,
        /*c_mod=*/(short)0, c, /*reuse_a=*/false, /*reuse_b=*/false);
}

// K index inside a 16x32 f16 fragment for half-slot h (0..15) and lane group g (0/1)
// per CDNA5 ISA 7.12.2: lanes 0-15 hold K {0..7,16..23}, lanes 16-31 hold K {8..15,24..31}
__device__ __forceinline__ int kmap(int h, int g) {
    return ((h & 8) << 1) + g * 8 + (h & 7);
}

// ------------------------------------------------------------------
// 1) per-point squared norms  norms[b*N+n] = sum_c x[b,c,n,0]^2
// ------------------------------------------------------------------
__global__ void k_rownorm(const float* __restrict__ x, float* __restrict__ norms) {
    int idx = blockIdx.x * blockDim.x + threadIdx.x;     // B*N
    int b = idx >> 10, n = idx & 1023;
    float s = 0.f;
    #pragma unroll 8
    for (int c = 0; c < CC; ++c) {
        float v = x[(((size_t)b * CC + c) * NN + n) * TT];
        s += v * v;
    }
    norms[idx] = s;
}

// ------------------------------------------------------------------
// 2) distance matrix via WMMA gram:  D = sqrt(max(ni+nj-2*P.P^T,0))
// ------------------------------------------------------------------
__global__ void k_gram_dist(const float* __restrict__ x, const float* __restrict__ norms,
                            float* __restrict__ D, float* __restrict__ dsum) {
    int b    = blockIdx.y;
    int wave = threadIdx.x >> 5;
    int lane = threadIdx.x & 31;
    int tile = blockIdx.x * 4 + wave;                    // 0..4095
    int ti = tile >> 6, tj = tile & 63;
    int i0 = ti * 16, j0 = tj * 16;
    int m = lane & 15, g = lane >> 4;

    v16h a, bf;
    #pragma unroll
    for (int h = 0; h < 16; ++h) {
        int c = kmap(h, g);                              // K = channel (C_IN = 32 exactly)
        a[h]  = (_Float16)x[(((size_t)b * CC + c) * NN + (i0 + m)) * TT];
        bf[h] = (_Float16)x[(((size_t)b * CC + c) * NN + (j0 + m)) * TT];
    }
    v8f acc = {};
    acc = wmma16(a, bf, acc);

    float* Db = D + (size_t)b * NN * NN;
    float tilesum = 0.f;
    #pragma unroll
    for (int r = 0; r < 8; ++r) {
        int i = i0 + r + 8 * g;
        int j = j0 + m;
        float d2 = norms[b * NN + i] + norms[b * NN + j] - 2.f * acc[r];
        float d = (d2 > 0.f) ? sqrtf(fmaxf(d2, 1e-12f)) : 0.f;
        Db[(size_t)i * NN + j] = d;
        tilesum += d;
    }
    #pragma unroll
    for (int off = 16; off > 0; off >>= 1) tilesum += __shfl_xor(tilesum, off, 32);
    if (lane == 0) atomicAdd(&dsum[b], tilesum);
}

// ------------------------------------------------------------------
// 3) per-row 8-NN selection + symmetric scatter into A
// ------------------------------------------------------------------
__global__ void k_topk_buildA(const float* __restrict__ D, const float* __restrict__ dsum,
                              float* __restrict__ A) {
    int wave = threadIdx.x >> 5;
    int lane = threadIdx.x & 31;
    int rg = blockIdx.x * 8 + wave;                      // 0..B*N-1
    int b = rg >> 10, i = rg & 1023;
    const float* row = D + (size_t)b * NN * NN + (size_t)i * NN;

    float locd[KNE]; int locj[KNE];
    #pragma unroll
    for (int s = 0; s < KNE; ++s) { locd[s] = 3.4e38f; locj[s] = -1; }

    for (int it = 0; it < NN / 32; ++it) {
        int j = lane + 32 * it;
        float d = row[j];
        if (d < locd[KNE - 1]) {
            locd[KNE - 1] = d; locj[KNE - 1] = j;
            #pragma unroll
            for (int s = KNE - 1; s >= 1; --s) {
                if (locd[s] < locd[s - 1]) {
                    float td = locd[s]; locd[s] = locd[s - 1]; locd[s - 1] = td;
                    int   tj = locj[s]; locj[s] = locj[s - 1]; locj[s - 1] = tj;
                }
            }
        }
    }

    float sg = dsum[b] * (1.f / ((float)NN * (float)NN));
    float inv_s2 = 1.f / (sg * sg);
    float* Ab = A + (size_t)b * NN * NN;

    for (int sel = 0; sel < KNE; ++sel) {
        float bd = locd[0]; int bj = locj[0];
        #pragma unroll
        for (int s = 1; s < KNE; ++s)
            if (locd[s] < bd || (locd[s] == bd && locj[s] < bj)) { bd = locd[s]; bj = locj[s]; }
        #pragma unroll
        for (int off = 16; off > 0; off >>= 1) {
            float od = __shfl_xor(bd, off, 32);
            int   oj = __shfl_xor(bj, off, 32);
            if (od < bd || (od == bd && oj < bj)) { bd = od; bj = oj; }
        }
        #pragma unroll
        for (int s = 0; s < KNE; ++s) if (locj[s] == bj) locd[s] = 3.4e38f;
        if (lane == sel && bj >= 0) {
            float w = 0.5f * expf(-bd * inv_s2);
            atomicAdd(&Ab[(size_t)i * NN + bj], w);
            atomicAdd(&Ab[(size_t)bj * NN + i], w);
        }
    }
}

// ------------------------------------------------------------------
// 4) degree then S = 2I - dinv*(diag(deg)-A)*dinv  stored as f16
// ------------------------------------------------------------------
__global__ void k_degree(const float* __restrict__ A, float* __restrict__ deg) {
    int wave = threadIdx.x >> 5, lane = threadIdx.x & 31;
    int rg = blockIdx.x * 8 + wave;
    int b = rg >> 10, i = rg & 1023;
    const float* row = A + (size_t)b * NN * NN + (size_t)i * NN;
    float s = 0.f;
    for (int j = lane; j < NN; j += 32) s += row[j];
    #pragma unroll
    for (int off = 16; off > 0; off >>= 1) s += __shfl_xor(s, off, 32);
    if (lane == 0) deg[rg] = s;
}

__global__ void k_buildS(const float* __restrict__ A, const float* __restrict__ deg,
                         _Float16* __restrict__ S) {
    size_t total = (size_t)BB * NN * NN;
    for (size_t idx = (size_t)blockIdx.x * blockDim.x + threadIdx.x; idx < total;
         idx += (size_t)gridDim.x * blockDim.x) {
        size_t r = idx & (size_t)(NN * NN - 1);
        int b = (int)(idx >> 20);
        int i = (int)(r >> 10), j = (int)(r & 1023);
        float di = deg[b * NN + i], dj = deg[b * NN + j];
        float dii = rsqrtf(di + 1e-6f), djj = rsqrtf(dj + 1e-6f);
        float s = dii * A[idx] * djj;
        if (i == j) s += 2.f - di * dii * dii;
        S[idx] = (_Float16)s;
    }
}

// ------------------------------------------------------------------
// 5) deterministic subspace init
// ------------------------------------------------------------------
__global__ void k_initY(float* __restrict__ Y) {
    int idx = blockIdx.x * blockDim.x + threadIdx.x;     // B*N*16
    unsigned h = (unsigned)idx * 2654435761u + 12345u;
    h ^= h >> 13; h *= 0x5bd1e995u; h ^= h >> 15;
    Y[idx] = (float)(h & 0xFFFF) * (1.f / 32768.f) - 1.f;
}

// ------------------------------------------------------------------
// 6) Z = S * Y  (N x N f16) x (N x 16) — WMMA with TDM-staged A panels
//    block = 8 waves; A panel [128 x 32] f16 DMA'd to LDS per K-step,
//    Y panel [32 x 16] staged once per K-step and shared by all waves.
// ------------------------------------------------------------------
__global__ void __launch_bounds__(256) k_gemmSY(const _Float16* __restrict__ S,
                                                const float* __restrict__ Yin,
                                                float* __restrict__ Zout) {
    __shared__ _Float16 As[128 * 32];   // 8 KB — first LDS allocation (lds_addr 0)
    __shared__ _Float16 Ys[32 * 16];    // 1 KB
    int b = blockIdx.y;
    int tid = threadIdx.x;
    int wave = tid >> 5, lane = tid & 31;
    int m = lane & 15, g = lane >> 4;
    int row0 = blockIdx.x * 128;
    const float* Yb = Yin + (size_t)b * NN * MS;
    v8f acc = {};

    for (int kt = 0; kt < NN / 32; ++kt) {
        int k0 = kt * 32;
        // stage Y K-panel (32 x 16) f32 -> f16
        #pragma unroll
        for (int r = 0; r < 2; ++r) {
            int e = tid + 256 * r;
            Ys[e] = (_Float16)Yb[(k0 + (e >> 4)) * MS + (e & 15)];
        }
#if __has_builtin(__builtin_amdgcn_tensor_load_to_lds)
        // ---- Tensor Data Mover: 2D tile 32 elems x 128 rows, stride N, 2B/elem ----
        if (tid < 32) {
            unsigned long long ga =
                (unsigned long long)(const void*)(S + (size_t)b * NN * NN +
                                                  (size_t)row0 * NN + k0);
            u32x4 g0;
            g0[0] = 1u;                                            // count=1 user D#
            g0[1] = 0u;                                            // lds_addr = 0 (As)
            g0[2] = (unsigned)(ga & 0xFFFFFFFFull);                // global_addr[31:0]
            g0[3] = (unsigned)((ga >> 32) & 0x1FFFFFFull) | (2u << 30); // addr|type=2
            i32x8 g1;
            g1[0] = (int)(1u << 16);          // data_size=2B, workgroup_mask=0
            g1[1] = (int)((unsigned)NN << 16);// tensor_dim0[15:0]
            g1[2] = (int)((unsigned)NN << 16);// tensor_dim0[31:16]=0 | tensor_dim1[15:0]
            g1[3] = (int)(32u << 16);         // tensor_dim1[31:16]=0 | tile_dim0=32
            g1[4] = (int)(128u);              // tile_dim1=128, tile_dim2=0
            g1[5] = (int)(NN);                // tensor_dim0_stride[31:0]
            g1[6] = 0;                        // stride0 hi | stride1 lo
            g1[7] = 0;                        // stride1 hi
            i32x4 z4 = {0, 0, 0, 0};
#if __clang_major__ >= 23
            i32x8 z8 = {0, 0, 0, 0, 0, 0, 0, 0};
            __builtin_amdgcn_tensor_load_to_lds(g0, g1, z4, z4, z8, 0);
#else
            __builtin_amdgcn_tensor_load_to_lds(g0, g1, z4, z4, 0);
#endif
#if __has_builtin(__builtin_amdgcn_s_wait_tensorcnt)
            __builtin_amdgcn_s_wait_tensorcnt(0);
#endif
        }
#else
        // fallback: cooperative synchronous copy (4096 halves / 256 threads)
        #pragma unroll
        for (int r = 0; r < 16; ++r) {
            int e = tid + 256 * r;
            As[e] = S[(size_t)b * NN * NN + (size_t)(row0 + (e >> 5)) * NN + k0 + (e & 31)];
        }
#endif
        __syncthreads();
        v16h a, bf;
        #pragma unroll
        for (int h = 0; h < 16; ++h) {
            int kk = kmap(h, g);
            a[h]  = As[(wave * 16 + m) * 32 + kk];
            bf[h] = Ys[kk * 16 + m];
        }
        acc = wmma16(a, bf, acc);
        __syncthreads();
    }
    int mt = blockIdx.x * 8 + wave;
    float* Zb = Zout + (size_t)b * NN * MS + (size_t)(mt * 16) * MS;
    #pragma unroll
    for (int r = 0; r < 8; ++r) Zb[(r + 8 * g) * MS + m] = acc[r];
}

// ------------------------------------------------------------------
// 7) modified Gram-Schmidt: Z -> orthonormal Y (one block per batch)
// ------------------------------------------------------------------
__global__ void k_gs(const float* __restrict__ Zin, float* __restrict__ Yout) {
    int b = blockIdx.x, tid = threadIdx.x;               // 256 threads
    const float* Z = Zin + (size_t)b * NN * MS;
    float* Y = Yout + (size_t)b * NN * MS;
    __shared__ float red[256];
    for (int k = 0; k < MS; ++k) {
        float v[4];
        #pragma unroll
        for (int r = 0; r < 4; ++r) v[r] = Z[(tid + 256 * r) * MS + k];
        for (int j = 0; j < k; ++j) {
            float yj[4]; float p = 0.f;
            #pragma unroll
            for (int r = 0; r < 4; ++r) { yj[r] = Y[(tid + 256 * r) * MS + j]; p += v[r] * yj[r]; }
            red[tid] = p; __syncthreads();
            for (int s = 128; s > 0; s >>= 1) { if (tid < s) red[tid] += red[tid + s]; __syncthreads(); }
            float dot = red[0]; __syncthreads();
            #pragma unroll
            for (int r = 0; r < 4; ++r) v[r] -= dot * yj[r];
        }
        float p = 0.f;
        #pragma unroll
        for (int r = 0; r < 4; ++r) p += v[r] * v[r];
        red[tid] = p; __syncthreads();
        for (int s = 128; s > 0; s >>= 1) { if (tid < s) red[tid] += red[tid + s]; __syncthreads(); }
        float inv = rsqrtf(red[0] + 1e-12f); __syncthreads();
        #pragma unroll
        for (int r = 0; r < 4; ++r) Y[(tid + 256 * r) * MS + k] = v[r] * inv;
        __syncthreads();
    }
}

// ------------------------------------------------------------------
// 8) Rayleigh-Ritz: H = Y^T Z (16x16), Jacobi eigensolve, basis = Y*V
// ------------------------------------------------------------------
__global__ void k_ritzH(const float* __restrict__ Y, const float* __restrict__ Z,
                        float* __restrict__ H) {
    int b = blockIdx.x, tid = threadIdx.x;               // 256 = 16x16
    int p = tid >> 4, q = tid & 15;
    const float* Yb = Y + (size_t)b * NN * MS;
    const float* Zb = Z + (size_t)b * NN * MS;
    float acc = 0.f;
    for (int n = 0; n < NN; ++n) acc += Yb[n * MS + p] * Zb[n * MS + q];
    H[b * 256 + tid] = acc;
}

__global__ void k_jacobi16(const float* __restrict__ H, float* __restrict__ V,
                           float* __restrict__ evals) {
    int b = blockIdx.x;
    if (threadIdx.x != 0) return;
    float a[MS][MS], v[MS][MS];
    for (int i = 0; i < MS; ++i)
        for (int j = 0; j < MS; ++j) {
            a[i][j] = 0.5f * (H[b * 256 + i * MS + j] + H[b * 256 + j * MS + i]);
            v[i][j] = (i == j) ? 1.f : 0.f;
        }
    for (int sweep = 0; sweep < 10; ++sweep) {
        for (int p = 0; p < MS - 1; ++p)
            for (int q = p + 1; q < MS; ++q) {
                float apq = a[p][q];
                if (fabsf(apq) < 1e-12f) continue;
                float theta = 0.5f * (a[q][q] - a[p][p]) / apq;
                float t = (theta >= 0.f ? 1.f : -1.f) / (fabsf(theta) + sqrtf(theta * theta + 1.f));
                float c = rsqrtf(t * t + 1.f), s = t * c;
                for (int k = 0; k < MS; ++k) {
                    float akp = a[k][p], akq = a[k][q];
                    a[k][p] = c * akp - s * akq;
                    a[k][q] = s * akp + c * akq;
                }
                for (int k = 0; k < MS; ++k) {
                    float apk = a[p][k], aqk = a[q][k];
                    a[p][k] = c * apk - s * aqk;
                    a[q][k] = s * apk + c * aqk;
                    float vkp = v[k][p], vkq = v[k][q];
                    v[k][p] = c * vkp - s * vkq;
                    v[k][q] = s * vkp + c * vkq;
                }
            }
    }
    int perm[MS];
    float lam[MS];
    for (int i = 0; i < MS; ++i) { perm[i] = i; lam[i] = a[i][i]; }
    for (int i = 0; i < MS - 1; ++i) {
        int best = i;
        for (int j = i + 1; j < MS; ++j) if (lam[perm[j]] > lam[perm[best]]) best = j;
        int t = perm[i]; perm[i] = perm[best]; perm[best] = t;
    }
    for (int k = 0; k < MS; ++k) {
        evals[b * MS + k] = lam[perm[k]];
        for (int i = 0; i < MS; ++i) V[b * 256 + i * MS + k] = v[i][perm[k]];
    }
}

__global__ void k_basis(const float* __restrict__ Y, const float* __restrict__ V,
                        float* __restrict__ basis) {
    __shared__ float Vs[256];
    int b = blockIdx.y;
    int n = blockIdx.x * 256 + threadIdx.x;
    Vs[threadIdx.x] = V[b * 256 + threadIdx.x];
    __syncthreads();
    float yr[MS];
    #pragma unroll
    for (int q = 0; q < MS; ++q) yr[q] = Y[(size_t)b * NN * MS + n * MS + q];
    #pragma unroll
    for (int k = 0; k < MS; ++k) {
        float s = 0.f;
        #pragma unroll
        for (int q = 0; q < MS; ++q) s += yr[q] * Vs[q * MS + k];
        basis[(size_t)b * NN * MS + n * MS + k] = s;
    }
}

// ------------------------------------------------------------------
// 9) forward spectral projection: xf[b][k][c*64+t] = sum_n basis[n][k] x[b,c,n,t]
//    (16 x N) x (N x 2048), K = 1024 — WMMA, shared A panel via LDS
// ------------------------------------------------------------------
__global__ void __launch_bounds__(256) k_fwd(const float* __restrict__ x,
                                             const float* __restrict__ basis,
                                             float* __restrict__ xf) {
    __shared__ _Float16 Bs[32 * 16];    // basis K-panel, shared by all 8 waves
    int b = blockIdx.y;
    int tid = threadIdx.x;
    int wave = tid >> 5, lane = tid & 31;
    int jt = blockIdx.x * 8 + wave;                      // 0..127
    int m = lane & 15, g = lane >> 4;
    int j = jt * 16 + m, c = j >> 6, t = j & 63;
    const float* Bb = basis + (size_t)b * NN * MS;
    v8f acc = {};
    for (int kt = 0; kt < NN / 32; ++kt) {
        int k0 = kt * 32;
        #pragma unroll
        for (int r = 0; r < 2; ++r) {
            int e = tid + 256 * r;
            Bs[e] = (_Float16)Bb[(k0 + (e >> 4)) * MS + (e & 15)];
        }
        __syncthreads();
        if (kt + 1 < NN / 32)
            __builtin_prefetch(&x[(((size_t)b * CC + c) * NN + (k0 + 32)) * TT + t], 0, 0);
        v16h a, bf;
        #pragma unroll
        for (int h = 0; h < 16; ++h) {
            int kk = kmap(h, g);
            a[h]  = Bs[kk * 16 + m];                     // A[k][n] = basis[n][k]
            bf[h] = (_Float16)x[(((size_t)b * CC + c) * NN + (k0 + kk)) * TT + t];
        }
        acc = wmma16(a, bf, acc);
        __syncthreads();
    }
    float* xfb = xf + (size_t)b * MS * (CC * TT);
    #pragma unroll
    for (int r = 0; r < 8; ++r) xfb[(r + 8 * g) * (CC * TT) + jt * 16 + m] = acc[r];
}

// ------------------------------------------------------------------
// 10) truncated rfft(4 bins) -> complex channel mix -> irfft(T=64)
// ------------------------------------------------------------------
__global__ void k_temporal(const float* __restrict__ xf, const float* __restrict__ Wr,
                           const float* __restrict__ Wi, float* __restrict__ os) {
    int kmode = blockIdx.x, b = blockIdx.y, t = threadIdx.x;   // 64 threads
    __shared__ float xs[CC * TT];
    __shared__ float Xre[CC * MT], Xim[CC * MT], Mre[CC * MT], Mim[CC * MT];
    const float* src = xf + ((size_t)b * MS + kmode) * (CC * TT);
    for (int r = 0; r < CC; ++r) xs[r * TT + t] = src[r * TT + t];
    __syncthreads();
    const float W0 = 6.28318530717958647692f / (float)TT;
    #pragma unroll
    for (int rep = 0; rep < 2; ++rep) {                  // 128 (i,m) items on 64 threads
        int item = t + 64 * rep;
        int i = item >> 2, mm = item & 3;
        float sre = 0.f, sim = 0.f;
        for (int tt = 0; tt < TT; ++tt) {
            float ang = W0 * (float)(mm * tt);
            float xv = xs[i * TT + tt];
            sre += xv * cosf(ang);
            sim -= xv * sinf(ang);
        }
        Xre[item] = sre; Xim[item] = sim;
    }
    __syncthreads();
    #pragma unroll
    for (int rep = 0; rep < 2; ++rep) {                  // 128 (o,m) items
        int item = t + 64 * rep;
        int o = item >> 2, mm = item & 3;
        float mre = 0.f, mim = 0.f;
        for (int i = 0; i < CC; ++i) {
            size_t widx = ((size_t)(i * CC + o) * MS + kmode) * MT + mm;
            float wr = Wr[widx], wi = Wi[widx];
            float xr = Xre[i * MT + mm], xi = Xim[i * MT + mm];
            mre += xr * wr - xi * wi;
            mim += xr * wi + xi * wr;
        }
        Mre[item] = mre; Mim[item] = mim;
    }
    __syncthreads();
    float* dst = os + ((size_t)b * MS + kmode) * (CC * TT);
    for (int o = 0; o < CC; ++o) {
        float acc = Mre[o * MT + 0];
        #pragma unroll
        for (int mm = 1; mm < MT; ++mm) {
            float ang = W0 * (float)(mm * t);
            acc += 2.f * (Mre[o * MT + mm] * cosf(ang) - Mim[o * MT + mm] * sinf(ang));
        }
        dst[o * TT + t] = acc * (1.f / (float)TT);
    }
}

// ------------------------------------------------------------------
// 11) back-projection: out[b,c,n,t] = sum_k basis[n][k] os[b][k][c*64+t]
//     (N x 16) x (16 x 2048), K padded 16->32 — single WMMA per tile
// ------------------------------------------------------------------
__global__ void k_out(const float* __restrict__ basis, const float* __restrict__ os,
                      float* __restrict__ out) {
    int b = blockIdx.y;
    int wave = threadIdx.x >> 5, lane = threadIdx.x & 31;
    int tile = blockIdx.x * 8 + wave;                    // 0..8191
    int mt = tile >> 7, jt = tile & 127;
    int m = lane & 15, g = lane >> 4;
    const float* Bb = basis + (size_t)b * NN * MS;
    const float* ob = os + (size_t)b * MS * (CC * TT);
    v16h a, bf;
    #pragma unroll
    for (int h = 0; h < 16; ++h) {
        int k = kmap(h, g);
        a[h]  = (k < MS) ? (_Float16)Bb[(mt * 16 + m) * MS + k] : (_Float16)0.0f;
        bf[h] = (k < MS) ? (_Float16)ob[(size_t)k * (CC * TT) + jt * 16 + m] : (_Float16)0.0f;
    }
    v8f acc = {};
    acc = wmma16(a, bf, acc);
    #pragma unroll
    for (int r = 0; r < 8; ++r) {
        int n = mt * 16 + r + 8 * g;
        int j = jt * 16 + m, c = j >> 6, t = j & 63;
        out[(((size_t)b * CC + c) * NN + n) * TT + t] = acc[r];
    }
}

// ------------------------------------------------------------------
extern "C" void kernel_launch(void* const* d_in, const int* in_sizes, int n_in,
                              void* d_out, int out_size, void* d_ws, size_t ws_size,
                              hipStream_t stream) {
    (void)in_sizes; (void)n_in; (void)out_size; (void)ws_size;
    const float* x  = (const float*)d_in[0];
    const float* Wr = (const float*)d_in[1];
    const float* Wi = (const float*)d_in[2];
    float* out = (float*)d_out;

    char* ws = (char*)d_ws;
    size_t off = 0;
    auto take = [&](size_t bytes) -> void* {
        void* p = ws + off;
        off = (off + bytes + 255) & ~(size_t)255;
        return p;
    };
    float*    D     = (float*)    take((size_t)BB * NN * NN * 4);
    float*    A     = (float*)    take((size_t)BB * NN * NN * 4);
    _Float16* S     = (_Float16*) take((size_t)BB * NN * NN * 2);
    float*    norms = (float*)    take((size_t)BB * NN * 4);
    float*    dsum  = (float*)    take((size_t)BB * 4);
    float*    deg   = (float*)    take((size_t)BB * NN * 4);
    float*    Y     = (float*)    take((size_t)BB * NN * MS * 4);
    float*    Z     = (float*)    take((size_t)BB * NN * MS * 4);
    float*    H     = (float*)    take((size_t)BB * 256 * 4);
    float*    V     = (float*)    take((size_t)BB * 256 * 4);
    float*    evals = (float*)    take((size_t)BB * MS * 4);
    float*    basis = (float*)    take((size_t)BB * NN * MS * 4);
    float*    xf    = (float*)    take((size_t)BB * MS * CC * TT * 4);
    float*    os    = (float*)    take((size_t)BB * MS * CC * TT * 4);

    hipMemsetAsync(A, 0, (size_t)BB * NN * NN * 4, stream);
    hipMemsetAsync(dsum, 0, (size_t)BB * 4, stream);

    k_rownorm    <<<dim3(BB * NN / 256), 256, 0, stream>>>(x, norms);
    k_gram_dist  <<<dim3(1024, BB), 128, 0, stream>>>(x, norms, D, dsum);
    k_topk_buildA<<<dim3(BB * NN / 8), 256, 0, stream>>>(D, dsum, A);
    k_degree     <<<dim3(BB * NN / 8), 256, 0, stream>>>(A, deg);
    k_buildS     <<<dim3(2048), 256, 0, stream>>>(A, deg, S);
    k_initY      <<<dim3(BB * NN * MS / 256), 256, 0, stream>>>(Y);

    for (int it = 0; it < 24; ++it) {
        k_gemmSY<<<dim3(8, BB), 256, 0, stream>>>(S, Y, Z);
        k_gs    <<<dim3(BB), 256, 0, stream>>>(Z, Y);
    }
    k_gemmSY  <<<dim3(8, BB), 256, 0, stream>>>(S, Y, Z);
    k_ritzH   <<<dim3(BB), 256, 0, stream>>>(Y, Z, H);
    k_jacobi16<<<dim3(BB), 64, 0, stream>>>(H, V, evals);
    k_basis   <<<dim3(NN / 256, BB), 256, 0, stream>>>(Y, V, basis);

    k_fwd     <<<dim3(16, BB), 256, 0, stream>>>(x, basis, xf);
    k_temporal<<<dim3(MS, BB), 64, 0, stream>>>(xf, Wr, Wi, os);
    k_out     <<<dim3(1024, BB), 256, 0, stream>>>(basis, os, out);
}
